// MHSA_29394756174281
// MI455X (gfx1250) — matrix operand order repaired and checked by reading, not compiled, not run on hardware
//
#include <hip/hip_runtime.h>

// Problem constants (match reference)
#define BB   8
#define NN   1024
#define NCC  64
#define CC   1024
#define HH   16
#define DD   64
#define LLEN 1088        // N + NC
#define QK_SCALE 0.125f  // 64^-0.5

typedef __attribute__((ext_vector_type(16))) __bf16 v16bf;
typedef __attribute__((ext_vector_type(8)))  __bf16 v8bf;
typedef __attribute__((ext_vector_type(8)))  float  v8f;

static __device__ __forceinline__ v16bf make_v16(v8bf lo, v8bf hi) {
  union { v16bf v; v8bf h[2]; } u;
  u.h[0] = lo; u.h[1] = hi;
  return u.v;
}
// A-operand (16x32 bf16): lane m = lane&15, half = lane>>4.
// elems 0..15 -> K = 16*half + e (contiguous)
static __device__ __forceinline__ v16bf load_a_frag(const __bf16* row, int half) {
  return make_v16(*(const v8bf*)(row + 16 * half),
                  *(const v8bf*)(row + 16 * half + 8));
}
// B-operand (32x16 bf16): lane n = lane&15 holds contiguous K = [16*half, +16)
static __device__ __forceinline__ v16bf load_b_frag(const __bf16* col) {
  return make_v16(*(const v8bf*)col, *(const v8bf*)(col + 8));
}
static __device__ __forceinline__ v8f wmma_bf16(v16bf a, v16bf b, v8f c) {
  return __builtin_amdgcn_wmma_f32_16x16x32_bf16(false, a, false, b, (short)0, c,
                                                 false, false);
}
// CDNA5 async global->LDS copy (16B per lane), tracked by ASYNCcnt.
static __device__ __forceinline__ void async_copy_b128(void* lds_dst,
                                                       const void* gsrc) {
  unsigned lds = (unsigned)(unsigned long long)lds_dst;  // low 32b = LDS offset
  asm volatile("global_load_async_to_lds_b128 %0, %1, off"
               :: "v"(lds), "v"(gsrc) : "memory");
}
static __device__ __forceinline__ void wait_async_le4() {
  asm volatile("s_wait_asynccnt 0x4" ::: "memory");
}
static __device__ __forceinline__ void wait_async_0() {
  asm volatile("s_wait_asynccnt 0x0" ::: "memory");
}

// ---------------------------------------------------------------- fp32 -> bf16
__global__ void cvt_f32_bf16_kernel(const float* __restrict__ in,
                                    __bf16* __restrict__ out, long n) {
  long i = (long)blockIdx.x * blockDim.x + threadIdx.x;
  long stride = (long)gridDim.x * blockDim.x;
  for (; i < n; i += stride) out[i] = (__bf16)in[i];
}

// ------------------------------------------------------------------- QKV GEMM
// out[m, n] = sum_k A[m,k] * W[n,k], n in [0, 3*C). Scatters to Q/K/V [B,H,L,D].
// grid = (Mrows/128, 3*C/128), block = 256 (8 waves: 4 along M x 2 along N).
// Double-buffered async global->LDS staging; steady-state loop unrolled x2 so
// the LDS buffer index is a literal at every site (no toggling address regs).
__global__ __launch_bounds__(256) void qkv_gemm_kernel(
    const __bf16* __restrict__ A, const __bf16* __restrict__ W,
    __bf16* __restrict__ Qo, __bf16* __restrict__ Ko, __bf16* __restrict__ Vo,
    int seqlen, int seqoff) {
  __shared__ __bf16 At[2][128][40];  // 80B row stride (16B aligned)
  __shared__ __bf16 Bt[2][128][40];
  const int tid = threadIdx.x;
  const int wave = tid >> 5, lane = tid & 31, half = lane >> 4, ln = lane & 15;
  const int wm = (wave & 3) * 32;   // wave's M offset within tile
  const int wn = (wave >> 2) * 64;  // wave's N offset within tile
  const int m0 = blockIdx.x * 128;
  const int n0 = blockIdx.y * 128;
  // per-thread staging slots: rows cr, cr+64; 8-elem chunk at column cc
  const int cr = tid >> 2;
  const int cc = (tid & 3) * 8;
  const __bf16* a_row0 = A + (size_t)(m0 + cr) * CC + cc;
  const __bf16* a_row1 = A + (size_t)(m0 + cr + 64) * CC + cc;
  const __bf16* w_row0 = W + (size_t)(n0 + cr) * CC + cc;
  const __bf16* w_row1 = W + (size_t)(n0 + cr + 64) * CC + cc;

  v8f acc[2][4] = {};
  auto issue = [&](int k0, int buf) {
    async_copy_b128(&At[buf][cr][cc],      a_row0 + k0);
    async_copy_b128(&At[buf][cr + 64][cc], a_row1 + k0);
    async_copy_b128(&Bt[buf][cr][cc],      w_row0 + k0);
    async_copy_b128(&Bt[buf][cr + 64][cc], w_row1 + k0);
  };
  auto compute = [&](int buf) {
    v16bf a0 = load_a_frag(&At[buf][wm + ln][0], half);
    v16bf a1 = load_a_frag(&At[buf][wm + 16 + ln][0], half);
#pragma unroll
    for (int t = 0; t < 4; ++t) {
      v16bf b = load_b_frag(&Bt[buf][wn + t * 16 + ln][16 * half]);
      acc[0][t] = wmma_bf16(a0, b, acc[0][t]);
      acc[1][t] = wmma_bf16(a1, b, acc[1][t]);
    }
  };

  const int NK = CC / 32;  // 32 (even)
  issue(0, 0);
  for (int kc = 0; kc < NK - 2; kc += 2) {
    issue((kc + 1) * 32, 1);
    wait_async_le4();
    __syncthreads();
    compute(0);
    __syncthreads();
    issue((kc + 2) * 32, 0);
    wait_async_le4();
    __syncthreads();
    compute(1);
    __syncthreads();
  }
  issue((NK - 1) * 32, 1);
  wait_async_le4();
  __syncthreads();
  compute(0);
  __syncthreads();
  wait_async_0();
  compute(1);

  // n0 multiple of 128 -> whole block shares which-of-qkv; head fixed per wave
  const int wsel = n0 >> 10;  // 0=q 1=k 2=v
  const int h = ((n0 + wn) >> 6) & (HH - 1);
  __bf16* dst = (wsel == 0) ? Qo : ((wsel == 1) ? Ko : Vo);
#pragma unroll
  for (int u = 0; u < 2; ++u) {
#pragma unroll
    for (int j = 0; j < 8; ++j) {
      int m = m0 + wm + u * 16 + j + 8 * half;  // C/D layout: row = j + 8*half
      int b = m / seqlen, l = m % seqlen;
      __bf16* orow = dst + (((size_t)b * HH + h) * LLEN + seqoff + l) * DD;
#pragma unroll
      for (int t = 0; t < 4; ++t)
        orow[t * 16 + ln] = (__bf16)acc[u][t][j];
    }
  }
}

// --------------------------------------------------------------- attention
// Flash attention over L=1088, D=64. grid = (L/64, B*H), block = 128 (4 waves).
__global__ __launch_bounds__(128) void attn_kernel(
    const __bf16* __restrict__ Q, const __bf16* __restrict__ K,
    const __bf16* __restrict__ V, __bf16* __restrict__ O) {
  __shared__ __bf16 Kt[32][80];      // [key][d], 160B row stride
  __shared__ __bf16 Vt[64][40];      // [d][key] (transposed), 80B row stride
  __shared__ __bf16 Pt[4][16][40];   // per-wave P re-stripe scratch
  const int tid = threadIdx.x;
  const int wave = tid >> 5, lane = tid & 31, half = lane >> 4, ln = lane & 15;
  const int bh = blockIdx.y;
  const int b = bh >> 4, h = bh & 15;
  const __bf16* Qb = Q + (size_t)bh * LLEN * DD;
  const __bf16* Kb = K + (size_t)bh * LLEN * DD;
  const __bf16* Vb = V + (size_t)bh * LLEN * DD;
  const int q0 = blockIdx.x * 64 + wave * 16;

  const __bf16* qrow = Qb + (size_t)(q0 + ln) * DD;
  v16bf qa0 = load_a_frag(qrow, half);        // d = 0..31
  v16bf qa1 = load_a_frag(qrow + 32, half);   // d = 32..63

  // per-thread staging slots: rows kr, kr+16; 8-elem chunk at column kcol
  const int kr = tid >> 3;          // 0..15
  const int kcol = (tid & 7) * 8;   // 0..56

  float mrow[8], lrow[8];
  v8f acc[4] = {};
#pragma unroll
  for (int j = 0; j < 8; ++j) { mrow[j] = -3.0e30f; lrow[j] = 0.0f; }

  for (int kc = 0; kc < LLEN / 32; ++kc) {
    const int kbase = kc * 32;
    // K tile via CDNA5 async copy (row-major [key][d])
    async_copy_b128(&Kt[kr][kcol],      Kb + (size_t)(kbase + kr) * DD + kcol);
    async_copy_b128(&Kt[kr + 16][kcol], Kb + (size_t)(kbase + kr + 16) * DD + kcol);
    // V tile: transpose into [d][key] for contiguous B-operand in P*V
    v8bf vv0 = *(const v8bf*)(Vb + (size_t)(kbase + kr) * DD + kcol);
    v8bf vv1 = *(const v8bf*)(Vb + (size_t)(kbase + kr + 16) * DD + kcol);
    // speculative prefetch of next tile (dropped silently if past the end)
    __builtin_prefetch(Vb + (size_t)(kbase + 32 + kr) * DD + kcol, 0, 1);
    __builtin_prefetch(Kb + (size_t)(kbase + 32 + kr) * DD + kcol, 0, 1);
#pragma unroll
    for (int e = 0; e < 8; ++e) {
      Vt[kcol + e][kr]      = vv0[e];
      Vt[kcol + e][kr + 16] = vv1[e];
    }
    wait_async_0();
    __syncthreads();

    // S = Q * K^T for this wave's 16 q rows x 32 keys
    v8f s0 = {}, s1 = {};
    {
      v16bf b00 = load_b_frag(&Kt[0 + ln][16 * half]);       // keys 0..15,  d 0..31
      v16bf b01 = load_b_frag(&Kt[0 + ln][32 + 16 * half]);  // keys 0..15,  d 32..63
      v16bf b10 = load_b_frag(&Kt[16 + ln][16 * half]);      // keys 16..31, d 0..31
      v16bf b11 = load_b_frag(&Kt[16 + ln][32 + 16 * half]); // keys 16..31, d 32..63
      s0 = wmma_bf16(qa0, b00, s0);
      s0 = wmma_bf16(qa1, b01, s0);
      s1 = wmma_bf16(qa0, b10, s1);
      s1 = wmma_bf16(qa1, b11, s1);
    }

    // Online softmax in C/D layout: row = j + 8*half across 16 lanes (cols)
#pragma unroll
    for (int j = 0; j < 8; ++j) {
      float v0 = s0[j] * QK_SCALE;
      float v1 = s1[j] * QK_SCALE;
      float mx = fmaxf(v0, v1);
      mx = fmaxf(mx, __shfl_xor(mx, 1, 16));
      mx = fmaxf(mx, __shfl_xor(mx, 2, 16));
      mx = fmaxf(mx, __shfl_xor(mx, 4, 16));
      mx = fmaxf(mx, __shfl_xor(mx, 8, 16));
      float mn = fmaxf(mrow[j], mx);
      float alpha = __expf(mrow[j] - mn);
      mrow[j] = mn;
      float p0 = __expf(v0 - mn);
      float p1 = __expf(v1 - mn);
      float ps = p0 + p1;
      ps += __shfl_xor(ps, 1, 16);
      ps += __shfl_xor(ps, 2, 16);
      ps += __shfl_xor(ps, 4, 16);
      ps += __shfl_xor(ps, 8, 16);
      lrow[j] = lrow[j] * alpha + ps;
      acc[0][j] *= alpha; acc[1][j] *= alpha;
      acc[2][j] *= alpha; acc[3][j] *= alpha;
      // spill P (C/D layout) to LDS for re-striping into A-operand layout
      Pt[wave][j + 8 * half][ln]      = (__bf16)p0;
      Pt[wave][j + 8 * half][16 + ln] = (__bf16)p1;
    }
    asm volatile("s_wait_dscnt 0x0" ::: "memory");  // wave-local LDS RAW fence
    v16bf pa = load_a_frag(&Pt[wave][ln][0], half); // P as A-operand, K=32 keys
#pragma unroll
    for (int t = 0; t < 4; ++t) {
      v16bf vb = load_b_frag(&Vt[t * 16 + ln][16 * half]);  // d cols t*16..+15
      acc[t] = wmma_bf16(pa, vb, acc[t]);
    }
    __syncthreads();
  }

  // epilogue: normalize and scatter to [B, L, H, D] bf16
#pragma unroll
  for (int j = 0; j < 8; ++j) {
    float inv = 1.0f / lrow[j];
    size_t m = (size_t)q0 + j + 8 * half;
    __bf16* orow = O + (((size_t)b * LLEN + m) * HH + h) * DD;
#pragma unroll
    for (int t = 0; t < 4; ++t)
      orow[t * 16 + ln] = (__bf16)(acc[t][j] * inv);
  }
}

// --------------------------------------------------------------- projection
// Out[m, n] = sum_k A[brow, k] * W[n, k] + bias[n] (fp32 out)
// A is attn-out [B, L, C] bf16; brow = b*L + seqoff + (m % seqlen)
// grid = (Mrows/128, C/128), block = 256 (8 waves: 4 x 2). Double-buffered,
// steady-state loop unrolled x2 (literal buffer indices).
__global__ __launch_bounds__(256) void proj_gemm_kernel(
    const __bf16* __restrict__ A, const __bf16* __restrict__ W,
    const float* __restrict__ bias, float* __restrict__ Out,
    int seqlen, int seqoff) {
  __shared__ __bf16 At[2][128][40];
  __shared__ __bf16 Bt[2][128][40];
  const int tid = threadIdx.x;
  const int wave = tid >> 5, lane = tid & 31, half = lane >> 4, ln = lane & 15;
  const int wm = (wave & 3) * 32;
  const int wn = (wave >> 2) * 64;
  const int m0 = blockIdx.x * 128;
  const int n0 = blockIdx.y * 128;
  const int cr = tid >> 2;
  const int cc = (tid & 3) * 8;
  // hoist row-index math (b = m/seqlen) out of the K loop
  int mA0 = m0 + cr, mA1 = m0 + cr + 64;
  const __bf16* a_row0 =
      A + ((size_t)(mA0 / seqlen) * LLEN + seqoff + (mA0 % seqlen)) * CC + cc;
  const __bf16* a_row1 =
      A + ((size_t)(mA1 / seqlen) * LLEN + seqoff + (mA1 % seqlen)) * CC + cc;
  const __bf16* w_row0 = W + (size_t)(n0 + cr) * CC + cc;
  const __bf16* w_row1 = W + (size_t)(n0 + cr + 64) * CC + cc;

  v8f acc[2][4] = {};
  auto issue = [&](int k0, int buf) {
    async_copy_b128(&At[buf][cr][cc],      a_row0 + k0);
    async_copy_b128(&At[buf][cr + 64][cc], a_row1 + k0);
    async_copy_b128(&Bt[buf][cr][cc],      w_row0 + k0);
    async_copy_b128(&Bt[buf][cr + 64][cc], w_row1 + k0);
  };
  auto compute = [&](int buf) {
    v16bf a0 = load_a_frag(&At[buf][wm + ln][0], half);
    v16bf a1 = load_a_frag(&At[buf][wm + 16 + ln][0], half);
#pragma unroll
    for (int t = 0; t < 4; ++t) {
      v16bf b = load_b_frag(&Bt[buf][wn + t * 16 + ln][16 * half]);
      acc[0][t] = wmma_bf16(a0, b, acc[0][t]);
      acc[1][t] = wmma_bf16(a1, b, acc[1][t]);
    }
  };

  const int NK = CC / 32;  // 32 (even)
  issue(0, 0);
  for (int kc = 0; kc < NK - 2; kc += 2) {
    issue((kc + 1) * 32, 1);
    wait_async_le4();
    __syncthreads();
    compute(0);
    __syncthreads();
    issue((kc + 2) * 32, 0);
    wait_async_le4();
    __syncthreads();
    compute(1);
    __syncthreads();
  }
  issue((NK - 1) * 32, 1);
  wait_async_le4();
  __syncthreads();
  compute(0);
  __syncthreads();
  wait_async_0();
  compute(1);

#pragma unroll
  for (int u = 0; u < 2; ++u) {
#pragma unroll
    for (int j = 0; j < 8; ++j) {
      int m = m0 + wm + u * 16 + j + 8 * half;
      float* orow = Out + (size_t)m * CC;
#pragma unroll
      for (int t = 0; t < 4; ++t) {
        int n = n0 + wn + t * 16 + ln;
        orow[n] = acc[u][t][j] + bias[n];
      }
    }
  }
}

// ------------------------------------------------------------------ launcher
extern "C" void kernel_launch(void* const* d_in, const int* in_sizes, int n_in,
                              void* d_out, int out_size, void* d_ws,
                              size_t ws_size, hipStream_t stream) {
  (void)in_sizes; (void)n_in; (void)out_size; (void)ws_size;
  const float* x      = (const float*)d_in[0];
  const float* cond   = (const float*)d_in[1];
  const float* Wqkv   = (const float*)d_in[2];
  const float* Wqkvc  = (const float*)d_in[3];
  const float* Wproj  = (const float*)d_in[4];
  const float* bproj  = (const float*)d_in[5];
  const float* Wprojc = (const float*)d_in[6];
  const float* bprojc = (const float*)d_in[7];
  float* out = (float*)d_out;

  char* wp = (char*)d_ws;
  auto alloc_bf = [&](size_t elems) {
    __bf16* p = (__bf16*)wp;
    wp += (elems * sizeof(__bf16) + 255) & ~(size_t)255;
    return p;
  };
  __bf16* x_bf     = alloc_bf((size_t)BB * NN * CC);
  __bf16* c_bf     = alloc_bf((size_t)BB * NCC * CC);
  __bf16* wqkv_bf  = alloc_bf((size_t)3 * CC * CC);
  __bf16* wqkvc_bf = alloc_bf((size_t)3 * CC * CC);
  __bf16* wpj_bf   = alloc_bf((size_t)CC * CC);
  __bf16* wpjc_bf  = alloc_bf((size_t)CC * CC);
  __bf16* q_bf     = alloc_bf((size_t)BB * HH * LLEN * DD);
  __bf16* k_bf     = alloc_bf((size_t)BB * HH * LLEN * DD);
  __bf16* v_bf     = alloc_bf((size_t)BB * HH * LLEN * DD);
  __bf16* at_bf    = alloc_bf((size_t)BB * LLEN * CC);

  cvt_f32_bf16_kernel<<<1024, 256, 0, stream>>>(x, x_bf, (long)BB * NN * CC);
  cvt_f32_bf16_kernel<<<512, 256, 0, stream>>>(cond, c_bf, (long)BB * NCC * CC);
  cvt_f32_bf16_kernel<<<1024, 256, 0, stream>>>(Wqkv, wqkv_bf, (long)3 * CC * CC);
  cvt_f32_bf16_kernel<<<1024, 256, 0, stream>>>(Wqkvc, wqkvc_bf, (long)3 * CC * CC);
  cvt_f32_bf16_kernel<<<512, 256, 0, stream>>>(Wproj, wpj_bf, (long)CC * CC);
  cvt_f32_bf16_kernel<<<512, 256, 0, stream>>>(Wprojc, wpjc_bf, (long)CC * CC);

  // QKV projections (x stream then cond stream, appended at seq offset N)
  qkv_gemm_kernel<<<dim3(BB * NN / 128, 3 * CC / 128), 256, 0, stream>>>(
      x_bf, wqkv_bf, q_bf, k_bf, v_bf, NN, 0);
  qkv_gemm_kernel<<<dim3(BB * NCC / 128, 3 * CC / 128), 256, 0, stream>>>(
      c_bf, wqkvc_bf, q_bf, k_bf, v_bf, NCC, NN);

  // Flash attention over joint sequence
  attn_kernel<<<dim3(LLEN / 64, BB * HH), 128, 0, stream>>>(q_bf, k_bf, v_bf,
                                                            at_bf);

  // Output projections (fp32 results, concatenated: x_out then cond_out)
  proj_gemm_kernel<<<dim3(BB * NN / 128, CC / 128), 256, 0, stream>>>(
      at_bf, wpj_bf, bproj, out, NN, 0);
  proj_gemm_kernel<<<dim3(BB * NCC / 128, CC / 128), 256, 0, stream>>>(
      at_bf, wpjc_bf, bprojc, out + (size_t)BB * NN * CC, NCC, NN);
}
